// EpipolarAttention_22643067584757
// MI455X (gfx1250) — compile-verified, matrix operand order checked by reading, not run on hardware
//
#include <hip/hip_runtime.h>
#include <hip/hip_bf16.h>

typedef __attribute__((ext_vector_type(16))) __bf16 v16bf;
typedef __attribute__((ext_vector_type(8)))  __bf16 v8bf;
typedef __attribute__((ext_vector_type(8)))  float  v8f;
typedef __attribute__((ext_vector_type(4)))  float  v4f;

#define BB    2
#define HW    4096
#define NTGT  4096
#define CCH   512
#define NH    8
#define KNN   32
#define DH    64
#define MROWS (BB * HW)          /* 8192 rows for src and for tgt (B*NTGT) */
#define ATT_SCALE 0.125f         /* DH^-0.5 = 64^-0.5 */

// ---------------------------------------------------------------------------
// fp32 -> bf16 straight convert, 8 elements/thread (2x b128 in, 1x b128 out)
// ---------------------------------------------------------------------------
__global__ __launch_bounds__(256) void cvt_bf16_kernel(const float* __restrict__ in,
                                                       __bf16* __restrict__ out, int n8) {
  int i = blockIdx.x * blockDim.x + threadIdx.x;
  if (i < n8) {
    v4f a = *(const v4f*)(in + (size_t)i * 8);
    v4f b = *(const v4f*)(in + (size_t)i * 8 + 4);
    v8bf o;
#pragma unroll
    for (int j = 0; j < 4; ++j) { o[j] = (__bf16)a[j]; o[j + 4] = (__bf16)b[j]; }
    *(v8bf*)(out + (size_t)i * 8) = o;
  }
}

// ---------------------------------------------------------------------------
// fp32 W[k][n] (512x512) -> bf16 Wt[n][k]  (so WMMA B-fragments are contiguous)
// ---------------------------------------------------------------------------
__global__ __launch_bounds__(256) void cvt_transpose_kernel(const float* __restrict__ W,
                                                            __bf16* __restrict__ Wt) {
  int i = blockIdx.x * blockDim.x + threadIdx.x;  // 0..512*512-1
  int k = i >> 9;
  int n = i & 511;
  Wt[(size_t)n * CCH + k] = (__bf16)W[i];
}

// ---------------------------------------------------------------------------
// bf16 WMMA GEMM:  out[M x 512] = A[M x 512] * Wt^T + bias
//   A row-major (M x K=512) bf16, Wt row-major (N=512 x K=512) bf16.
//   One wave -> 16x64 strip (4 WMMA tiles, A-fragment reused 4x per K-step).
//   8 waves/block -> 16 x 512 strip == full N; grid.x = M/16.
//   Fragment layout per CDNA5 ISA (16-bit A 16x32): lanes 0-15 hold K
//   {0..7,16..23}, lanes 16-31 hold K {8..15,24..31} -> two b128 loads each.
// ---------------------------------------------------------------------------
template <int STORE_F32>
__global__ __launch_bounds__(256) void gemm_bf16_wmma(const __bf16* __restrict__ A,
                                                      const __bf16* __restrict__ Bt,
                                                      const float* __restrict__ bias,
                                                      float* __restrict__ outF,
                                                      __bf16* __restrict__ outH) {
  const int lane  = threadIdx.x & 31;
  const int wave  = threadIdx.x >> 5;
  const int tileM = blockIdx.x << 4;
  const int tileN = wave << 6;           // 64 columns per wave
  const int half  = lane & 15;
  const int kOff  = (lane < 16) ? 0 : 8;

  const __bf16* ap = A  + (size_t)(tileM + half) * CCH + kOff;
  const __bf16* bp = Bt + (size_t)(tileN + half) * CCH + kOff;

  v8f acc0 = {}, acc1 = {}, acc2 = {}, acc3 = {};
#pragma unroll
  for (int k0 = 0; k0 < CCH; k0 += 32) {
    v8bf alo = *(const v8bf*)(ap + k0);        // K = k0+kOff .. +7
    v8bf ahi = *(const v8bf*)(ap + k0 + 16);   // K = k0+kOff+16 .. +23
    v16bf a = __builtin_shufflevector(alo, ahi, 0,1,2,3,4,5,6,7,8,9,10,11,12,13,14,15);

    v8bf b0l = *(const v8bf*)(bp + k0);
    v8bf b0h = *(const v8bf*)(bp + k0 + 16);
    acc0 = __builtin_amdgcn_wmma_f32_16x16x32_bf16(false, a, false,
             __builtin_shufflevector(b0l, b0h, 0,1,2,3,4,5,6,7,8,9,10,11,12,13,14,15),
             (short)0, acc0, false, false);

    v8bf b1l = *(const v8bf*)(bp + 16 * CCH + k0);
    v8bf b1h = *(const v8bf*)(bp + 16 * CCH + k0 + 16);
    acc1 = __builtin_amdgcn_wmma_f32_16x16x32_bf16(false, a, false,
             __builtin_shufflevector(b1l, b1h, 0,1,2,3,4,5,6,7,8,9,10,11,12,13,14,15),
             (short)0, acc1, false, false);

    v8bf b2l = *(const v8bf*)(bp + 32 * CCH + k0);
    v8bf b2h = *(const v8bf*)(bp + 32 * CCH + k0 + 16);
    acc2 = __builtin_amdgcn_wmma_f32_16x16x32_bf16(false, a, false,
             __builtin_shufflevector(b2l, b2h, 0,1,2,3,4,5,6,7,8,9,10,11,12,13,14,15),
             (short)0, acc2, false, false);

    v8bf b3l = *(const v8bf*)(bp + 48 * CCH + k0);
    v8bf b3h = *(const v8bf*)(bp + 48 * CCH + k0 + 16);
    acc3 = __builtin_amdgcn_wmma_f32_16x16x32_bf16(false, a, false,
             __builtin_shufflevector(b3l, b3h, 0,1,2,3,4,5,6,7,8,9,10,11,12,13,14,15),
             (short)0, acc3, false, false);
  }

  // Epilogue: C/D layout -> lanes 0-15: rows r, N=lane; lanes 16-31: rows 8+r.
  const int rb = tileM + ((lane < 16) ? 0 : 8);
  v8f accs[4] = {acc0, acc1, acc2, acc3};
#pragma unroll
  for (int t = 0; t < 4; ++t) {
    const int   col = tileN + (t << 4) + half;
    const float bv  = bias[col];
#pragma unroll
    for (int r = 0; r < 8; ++r) {
      float v = accs[t][r] + bv;
      size_t off = (size_t)(rb + r) * CCH + col;
      if (STORE_F32) outF[off] = v;
      else           outH[off] = (__bf16)v;
    }
  }
}

// ---------------------------------------------------------------------------
// Gather-attention core. Block = one query position (b*HW+q); wave = head h;
// lane = neighbor kn. K/V are bf16 and L2-resident (8 MB each).
// ---------------------------------------------------------------------------
__global__ __launch_bounds__(256) void epi_attn_kernel(const __bf16* __restrict__ Q,
                                                       const __bf16* __restrict__ Kc,
                                                       const __bf16* __restrict__ V,
                                                       const int* __restrict__ idx,
                                                       const float* __restrict__ wts,
                                                       __bf16* __restrict__ AO) {
  const int pos  = blockIdx.x;        // b*HW + q
  const int b    = pos >> 12;         // / 4096
  const int h    = threadIdx.x >> 5;
  const int lane = threadIdx.x & 31;

  const __bf16* qrow  = Q + (size_t)pos * CCH + h * DH;
  const int     myIdx = idx[(size_t)pos * KNN + lane];
  const __bf16* krow  = Kc + (size_t)(b * NTGT + myIdx) * CCH + h * DH;

  // logits[h][lane] = <q_h, k_h(idx[lane])>
  float dot = 0.f;
#pragma unroll
  for (int d = 0; d < DH; d += 8) {
    v8bf qv = *(const v8bf*)(qrow + d);
    v8bf kv = *(const v8bf*)(krow + d);
#pragma unroll
    for (int j = 0; j < 8; ++j) dot += (float)qv[j] * (float)kv[j];
  }
  float logit = dot * ATT_SCALE + wts[(size_t)pos * KNN + lane];

  // softmax across the 32 lanes of the wave
  float m = logit;
#pragma unroll
  for (int off = 16; off > 0; off >>= 1) m = fmaxf(m, __shfl_xor(m, off, 32));
  float e = __expf(logit - m);
  float s = e;
#pragma unroll
  for (int off = 16; off > 0; off >>= 1) s += __shfl_xor(s, off, 32);
  const float attn = e / s;

  // out[h][d] = sum_kn attn[kn] * V[idx[kn]][h*64+d]; lanes cover d and d+32.
  float o0 = 0.f, o1 = 0.f;
#pragma unroll 4
  for (int kn = 0; kn < KNN; ++kn) {
    float a  = __shfl(attn, kn, 32);
    int   id = __shfl(myIdx, kn, 32);
    const __bf16* vrow = V + (size_t)(b * NTGT + id) * CCH + h * DH;
    o0 += a * (float)vrow[lane];
    o1 += a * (float)vrow[lane + 32];
  }
  __bf16* orow = AO + (size_t)pos * CCH + h * DH;
  orow[lane]      = (__bf16)o0;
  orow[lane + 32] = (__bf16)o1;
}

// ---------------------------------------------------------------------------
// Launcher
// ---------------------------------------------------------------------------
extern "C" void kernel_launch(void* const* d_in, const int* in_sizes, int n_in,
                              void* d_out, int out_size, void* d_ws, size_t ws_size,
                              hipStream_t stream) {
  (void)in_sizes; (void)n_in; (void)out_size; (void)ws_size;

  const float* src = (const float*)d_in[0];
  const float* tgt = (const float*)d_in[1];
  const int*   idx = (const int*)d_in[2];
  const float* wts = (const float*)d_in[3];
  const float* Wq  = (const float*)d_in[4];
  const float* bq  = (const float*)d_in[5];
  const float* Wk  = (const float*)d_in[6];
  const float* bk  = (const float*)d_in[7];
  const float* Wv  = (const float*)d_in[8];
  const float* bv  = (const float*)d_in[9];
  const float* Wo  = (const float*)d_in[10];
  const float* bo  = (const float*)d_in[11];
  float* out = (float*)d_out;

  // Workspace layout (bf16 staging, ~52 MB total)
  const size_t nAct = (size_t)MROWS * CCH;   // 4,194,304 elements
  const size_t nW   = (size_t)CCH * CCH;     //   262,144 elements
  char* p = (char*)d_ws;
  __bf16* srcH = (__bf16*)p; p += nAct * 2;
  __bf16* tgtH = (__bf16*)p; p += nAct * 2;
  __bf16* WqT  = (__bf16*)p; p += nW * 2;
  __bf16* WkT  = (__bf16*)p; p += nW * 2;
  __bf16* WvT  = (__bf16*)p; p += nW * 2;
  __bf16* WoT  = (__bf16*)p; p += nW * 2;
  __bf16* Qh   = (__bf16*)p; p += nAct * 2;
  __bf16* Kh   = (__bf16*)p; p += nAct * 2;
  __bf16* Vh   = (__bf16*)p; p += nAct * 2;
  __bf16* AOh  = (__bf16*)p; p += nAct * 2;

  const dim3 blk(256);

  // Stage 0: precision staging (8 elems/thread)
  const unsigned n8 = (unsigned)(nAct / 8);
  cvt_bf16_kernel<<<dim3((n8 + 255) / 256), blk, 0, stream>>>(src, srcH, (int)n8);
  cvt_bf16_kernel<<<dim3((n8 + 255) / 256), blk, 0, stream>>>(tgt, tgtH, (int)n8);
  cvt_transpose_kernel<<<dim3((unsigned)(nW / 256)), blk, 0, stream>>>(Wq, WqT);
  cvt_transpose_kernel<<<dim3((unsigned)(nW / 256)), blk, 0, stream>>>(Wk, WkT);
  cvt_transpose_kernel<<<dim3((unsigned)(nW / 256)), blk, 0, stream>>>(Wv, WvT);
  cvt_transpose_kernel<<<dim3((unsigned)(nW / 256)), blk, 0, stream>>>(Wo, WoT);

  // Stage 1: Q/K/V projections (bf16 WMMA, bf16 out). Grid: M/16 blocks,
  // 8 waves x 64 cols = full N=512 per block.
  const dim3 ggrid(MROWS / 16);
  gemm_bf16_wmma<0><<<ggrid, blk, 0, stream>>>(srcH, WqT, bq, nullptr, Qh);
  gemm_bf16_wmma<0><<<ggrid, blk, 0, stream>>>(tgtH, WkT, bk, nullptr, Kh);
  gemm_bf16_wmma<0><<<ggrid, blk, 0, stream>>>(tgtH, WvT, bv, nullptr, Vh);

  // Stage 2: KNN gather attention (L2-resident bf16 K/V)
  epi_attn_kernel<<<dim3(MROWS), blk, 0, stream>>>(Qh, Kh, Vh, idx, wts, AOh);

  // Stage 3: output projection (fp32 out)
  gemm_bf16_wmma<1><<<ggrid, blk, 0, stream>>>(AOh, WoT, bo, out, nullptr);
}
